// Model_58557584114085
// MI455X (gfx1250) — compile-verified, matrix operand order checked by reading, not run on hardware
//
#include <hip/hip_runtime.h>
#include <hip/hip_bf16.h>
#include <stdint.h>
#include <stddef.h>

// ---------------------------------------------------------------------------
// GNN contrastive model for MI455X (gfx1250).
//  - All dense math (2x5 MLP layers x2 branches, projector, 4096x4096 logits)
//    runs through a bf16 WMMA GEMM (v_wmma_f32_16x16x32_bf16, f32 accum).
//    Block tile 128x64, 4 waves (2x2), wave tile 64x32 = 4x2 fragments ->
//    8 WMMAs per 6 fragment loads per k-step, 8 independent acc chains.
//  - Message passing uses f32 global atomics (agg buffer ~128MB, L2-resident).
//  - Dims padded: EMB 300->320, hidden 600->640, rows 100000->100096.
// ---------------------------------------------------------------------------

#define EMB   300
#define CP    320     // padded EMB    (10 * 32)
#define HID   600
#define HP    640     // padded hidden (20 * 32, 10 * 64)
#define NLAY  5
#define NG    4096
#define BN_EPS 1e-5f

typedef __attribute__((ext_vector_type(16))) __bf16 v16bf;
typedef __attribute__((ext_vector_type(8)))  float  v8f;

__device__ __forceinline__ unsigned short f2bf(float f) {
  union { float f; unsigned u; } x; x.f = f;
  unsigned r = x.u + 0x7fffu + ((x.u >> 16) & 1u);   // round-to-nearest-even
  return (unsigned short)(r >> 16);
}

// ---------------- embedding: h[i,c] = atom_emb1[x0] + atom_emb2[x1] --------
__global__ void embed_kernel(const int* __restrict__ x,
                             const float* __restrict__ ae1,
                             const float* __restrict__ ae2,
                             float* __restrict__ h, int M) {
  int idx = blockIdx.x * blockDim.x + threadIdx.x;
  int i = idx / CP, c = idx % CP;
  if (i >= M) return;
  float v = 0.f;
  if (c < EMB)
    v = ae1[(size_t)x[2 * i] * EMB + c] + ae2[(size_t)x[2 * i + 1] * EMB + c];
  h[(size_t)i * CP + c] = v;
}

// ---------------- edge messages + scatter-add into agg ---------------------
__global__ void edge_scatter(const float* __restrict__ h,
                             const int* __restrict__ ei,
                             const int* __restrict__ ea,
                             const float* __restrict__ ee1,
                             const float* __restrict__ ee2,
                             float* __restrict__ agg, int E) {
  int idx = blockIdx.x * blockDim.x + threadIdx.x;
  int e = idx / 75;
  int c = (idx % 75) * 4;            // 75 * 4 = 300 valid columns
  if (e >= E) return;
  int s  = ei[e];
  int d  = ei[E + e];
  int a0 = ea[2 * e];
  int a1 = ea[2 * e + 1];
  const float* hp = h   + (size_t)s  * CP  + c;
  const float* e1 = ee1 + (size_t)a0 * EMB + c;
  const float* e2 = ee2 + (size_t)a1 * EMB + c;
  float* ap       = agg + (size_t)d  * CP  + c;
#pragma unroll
  for (int j = 0; j < 4; ++j)
    atomicAdd(&ap[j], hp[j] + e1[j] + e2[j]);
}

// --------- self-loop: agg += h + self_emb ; also emit bf16 copy ------------
__global__ void add_self_bf(const float* __restrict__ h,
                            const float* __restrict__ ee1,
                            const float* __restrict__ ee2,
                            float* __restrict__ agg,
                            unsigned short* __restrict__ aggbf, int M) {
  int idx = blockIdx.x * blockDim.x + threadIdx.x;
  int i = idx / CP, c = idx % CP;
  if (i >= M) return;
  size_t o = (size_t)i * CP + c;
  float v = 0.f;
  if (c < EMB)
    v = agg[o] + h[o] + ee1[4 * EMB + c] + ee2[c];   // self edge_attr = [4,0]
  agg[o]   = v;
  aggbf[o] = f2bf(v);
}

// ---------------- WMMA GEMM: C = op(A[M,K] * Bt[N,K]^T + bias) -------------
// Block tile 128x64: 4 waves arranged 2x2; each wave computes 64x32 via
// 4x2 fragments of v_wmma_f32_16x16x32_bf16. A/Bt fragments are 16
// contiguous bf16 per lane (k = 16*(lane/16)+i, same order for A and Bt) -> two b128 loads.
__global__ void __launch_bounds__(128)
wmma_gemm(const unsigned short* __restrict__ A,
          const unsigned short* __restrict__ Bt,
          const float* __restrict__ bias,
          float* __restrict__ Cf,
          unsigned short* __restrict__ Cbf,
          int Mstore, int N, int K, float scale, int relu) {
  const int lane = threadIdx.x & 31;
  const int wave = threadIdx.x >> 5;
  const int m0 = blockIdx.y * 128 + (wave >> 1) * 64;
  const int n0 = blockIdx.x * 64  + (wave & 1) * 32;
  const int lr = lane & 15;             // row (A) / col (B) within fragment
  const int lk = (lane >> 4) << 4;      // k sub-offset: lanes 16-31 hold K+16

  const unsigned short* ap = A  + (size_t)(m0 + lr) * K + lk;
  const unsigned short* bp = Bt + (size_t)(n0 + lr) * K + lk;
  const size_t rowstep = (size_t)16 * K;

  v8f acc[4][2] = {};
  for (int k = 0; k < K; k += 32) {
    __builtin_prefetch(ap + k + 256, 0, 1);   // global_prefetch_b8 (A stream)
    v16bf b0 = *(const v16bf*)(bp + k);
    v16bf b1 = *(const v16bf*)(bp + rowstep + k);
#pragma unroll
    for (int mf = 0; mf < 4; ++mf) {
      v16bf a = *(const v16bf*)(ap + (size_t)mf * rowstep + k);
      acc[mf][0] = __builtin_amdgcn_wmma_f32_16x16x32_bf16(false, a, false, b0,
                       (short)0, acc[mf][0], false, false);
      acc[mf][1] = __builtin_amdgcn_wmma_f32_16x16x32_bf16(false, a, false, b1,
                       (short)0, acc[mf][1], false, false);
    }
  }

  // C/D layout: lane -> n = lane%16, m = (lane/16)*8 + vgpr_index
  const int cn = lane & 15;
  const int mb = (lane >> 4) << 3;
#pragma unroll
  for (int mf = 0; mf < 4; ++mf) {
#pragma unroll
    for (int nf = 0; nf < 2; ++nf) {
      const int ncol = n0 + nf * 16 + cn;
      const float bv = bias ? bias[ncol] : 0.f;
#pragma unroll
      for (int r = 0; r < 8; ++r) {
        const int row = m0 + mf * 16 + mb + r;
        if (row < Mstore) {
          float v = (acc[mf][nf][r] + bv) * scale;
          if (relu) v = fmaxf(v, 0.f);
          if (Cf)  Cf [(size_t)row * N + ncol] = v;
          if (Cbf) Cbf[(size_t)row * N + ncol] = f2bf(v);
        }
      }
    }
  }
}

// ---------------- batch-norm over nodes ------------------------------------
__global__ void bn_stats(const float* __restrict__ h,
                         float* __restrict__ sums, float* __restrict__ sqs,
                         int M) {
  int c = threadIdx.x;     // blockDim.x == CP
  float s = 0.f, q = 0.f;
  for (int i = blockIdx.x; i < M; i += gridDim.x) {
    float v = (c < EMB) ? h[(size_t)i * CP + c] : 0.f;
    s += v; q += v * v;
  }
  if (c < EMB) { atomicAdd(&sums[c], s); atomicAdd(&sqs[c], q); }
}

__global__ void bn_apply(float* __restrict__ h, unsigned short* __restrict__ hbf,
                         const float* __restrict__ sums, const float* __restrict__ sqs,
                         const float* __restrict__ gamma, const float* __restrict__ beta,
                         int M, int relu) {
  int idx = blockIdx.x * blockDim.x + threadIdx.x;
  int i = idx / CP, c = idx % CP;
  if (i >= M) return;
  float v = 0.f;
  if (c < EMB) {
    float mean = sums[c] / (float)M;
    float var  = sqs[c] / (float)M - mean * mean;
    float rstd = rsqrtf(var + BN_EPS);
    v = (h[(size_t)i * CP + c] - mean) * rstd * gamma[c] + beta[c];
    if (relu) v = fmaxf(v, 0.f);
  }
  h[(size_t)i * CP + c]   = v;
  hbf[(size_t)i * CP + c] = f2bf(v);
}

// ---------------- graph pooling + L2 normalize -----------------------------
__global__ void pool_sum(const float* __restrict__ src, const int* __restrict__ batch,
                         float* __restrict__ pooled, int M) {
  int idx = blockIdx.x * blockDim.x + threadIdx.x;
  int i = idx / 75;
  int c = (idx % 75) * 4;
  if (i >= M) return;
  int g = batch[i];
  const float* s = src + (size_t)i * CP + c;
  float* p = pooled + (size_t)g * CP + c;
#pragma unroll
  for (int j = 0; j < 4; ++j) atomicAdd(&p[j], s[j]);
}

__global__ void pool_count(const int* __restrict__ batch, float* __restrict__ cnt, int M) {
  int i = blockIdx.x * blockDim.x + threadIdx.x;
  if (i < M) atomicAdd(&cnt[batch[i]], 1.f);
}

__global__ void pool_norm(const float* __restrict__ pooled, const float* __restrict__ cnt,
                          unsigned short* __restrict__ fbf) {
  __shared__ float sh[CP];
  __shared__ float nrm;
  int g = blockIdx.x, c = threadIdx.x;
  float mean = (c < EMB) ? pooled[(size_t)g * CP + c] / fmaxf(cnt[g], 1.f) : 0.f;
  sh[c] = mean * mean;
  __syncthreads();
  if (c == 0) {
    float s = 0.f;
    for (int i = 0; i < EMB; ++i) s += sh[i];
    nrm = fmaxf(sqrtf(s), 1e-12f);
  }
  __syncthreads();
  fbf[(size_t)g * CP + c] = f2bf(mean / nrm);
}

// ---------------- weight prep: transpose + pad + bf16 ----------------------
__global__ void prep_w1(const float* __restrict__ w1, unsigned short* __restrict__ w1t) {
  int idx = blockIdx.x * blockDim.x + threadIdx.x;
  if (idx >= NLAY * HP * CP) return;
  int l = idx / (HP * CP), rem = idx % (HP * CP);
  int n = rem / CP, k = rem % CP;   // Bt layout: [N][K]
  float v = (n < HID && k < EMB) ? w1[(size_t)l * EMB * HID + (size_t)k * HID + n] : 0.f;
  w1t[idx] = f2bf(v);
}

__global__ void prep_w2(const float* __restrict__ w2, unsigned short* __restrict__ w2t) {
  int idx = blockIdx.x * blockDim.x + threadIdx.x;
  if (idx >= NLAY * CP * HP) return;
  int l = idx / (CP * HP), rem = idx % (CP * HP);
  int n = rem / HP, k = rem % HP;
  float v = (n < EMB && k < HID) ? w2[(size_t)l * HID * EMB + (size_t)k * EMB + n] : 0.f;
  w2t[idx] = f2bf(v);
}

__global__ void prep_proj(const float* __restrict__ pw, unsigned short* __restrict__ pjt) {
  int idx = blockIdx.x * blockDim.x + threadIdx.x;
  if (idx >= CP * CP) return;
  int n = idx / CP, k = idx % CP;
  float v = (n < EMB && k < EMB) ? pw[(size_t)k * EMB + n] : 0.f;
  pjt[idx] = f2bf(v);
}

__global__ void pad_bias(const float* __restrict__ src, float* __restrict__ dst,
                         int levels, int sn, int dn) {
  int idx = blockIdx.x * blockDim.x + threadIdx.x;
  if (idx >= levels * dn) return;
  int l = idx / dn, n = idx % dn;
  dst[idx] = (n < sn) ? src[(size_t)l * sn + n] : 0.f;
}

// ---------------------------------------------------------------------------
static inline char* ws_take(char*& p, size_t bytes) {
  char* r = p;
  p += (bytes + 255) & ~(size_t)255;
  return r;
}

static inline void launch_gemm(const unsigned short* A, const unsigned short* Bt,
                               const float* bias, float* Cf, unsigned short* Cbf,
                               int Mstore, int Mpad, int N, int K,
                               float scale, int relu, hipStream_t s) {
  dim3 grid(N / 64, Mpad / 128);
  wmma_gemm<<<grid, 128, 0, s>>>(A, Bt, bias, Cf, Cbf, Mstore, N, K, scale, relu);
}

extern "C" void kernel_launch(void* const* d_in, const int* in_sizes, int n_in,
                              void* d_out, int out_size, void* d_ws, size_t ws_size,
                              hipStream_t stream) {
  (void)n_in; (void)out_size; (void)ws_size;
  const int M  = in_sizes[0] / 2;          // 100000 nodes
  const int E  = in_sizes[1] / 2;          // 400000 edges
  const int MP = (M + 127) & ~127;         // row-padded for 128-row GEMM tiles

  const int* xs[2]  = { (const int*)d_in[0], (const int*)d_in[4] };
  const int* eis[2] = { (const int*)d_in[1], (const int*)d_in[5] };
  const int* eas[2] = { (const int*)d_in[2], (const int*)d_in[6] };
  const int* bts[2] = { (const int*)d_in[3], (const int*)d_in[7] };
  const float* ae1   = (const float*)d_in[8];
  const float* ae2   = (const float*)d_in[9];
  const float* ee1   = (const float*)d_in[10];   // [5,6,300]
  const float* ee2   = (const float*)d_in[11];   // [5,3,300]
  const float* w1    = (const float*)d_in[12];
  const float* b1    = (const float*)d_in[13];
  const float* w2    = (const float*)d_in[14];
  const float* b2    = (const float*)d_in[15];
  const float* gam   = (const float*)d_in[16];
  const float* bet   = (const float*)d_in[17];
  const float* pw    = (const float*)d_in[18];
  const float* pb    = (const float*)d_in[19];

  // ---- workspace carve (~395 MB; mid_bf aliases agg, live ranges disjoint)
  char* p = (char*)d_ws;
  float*          h      = (float*)         ws_take(p, (size_t)MP * CP * 4);
  float*          agg    = (float*)         ws_take(p, (size_t)MP * HP * 2); // == MP*CP*4 bytes
  unsigned short* midbf  = (unsigned short*)agg;                             // alias
  unsigned short* aggbf  = (unsigned short*)ws_take(p, (size_t)MP * CP * 2);
  unsigned short* hbf    = (unsigned short*)ws_take(p, (size_t)MP * CP * 2);
  float*          pooled = (float*)         ws_take(p, ((size_t)NG * CP + NG) * 4);
  float*          cnt    = pooled + (size_t)NG * CP;
  float*          stats  = (float*)         ws_take(p, (size_t)2 * CP * 4);
  float*          sums   = stats;
  float*          sqs    = stats + CP;
  unsigned short* fbf0   = (unsigned short*)ws_take(p, (size_t)NG * CP * 2);
  unsigned short* fbf1   = (unsigned short*)ws_take(p, (size_t)NG * CP * 2);
  unsigned short* w1t    = (unsigned short*)ws_take(p, (size_t)NLAY * HP * CP * 2);
  unsigned short* w2t    = (unsigned short*)ws_take(p, (size_t)NLAY * CP * HP * 2);
  unsigned short* pjt    = (unsigned short*)ws_take(p, (size_t)CP * CP * 2);
  float*          b1p    = (float*)         ws_take(p, (size_t)NLAY * HP * 4);
  float*          b2p    = (float*)         ws_take(p, (size_t)NLAY * CP * 4);
  float*          pbp    = (float*)         ws_take(p, (size_t)CP * 4);

  // ---- weight prep (cheap, deterministic, done every call)
  {
    int t1 = NLAY * HP * CP;
    prep_w1<<<(t1 + 255) / 256, 256, 0, stream>>>(w1, w1t);
    int t2 = NLAY * CP * HP;
    prep_w2<<<(t2 + 255) / 256, 256, 0, stream>>>(w2, w2t);
    prep_proj<<<(CP * CP + 255) / 256, 256, 0, stream>>>(pw, pjt);
    pad_bias<<<(NLAY * HP + 255) / 256, 256, 0, stream>>>(b1, b1p, NLAY, HID, HP);
    pad_bias<<<(NLAY * CP + 255) / 256, 256, 0, stream>>>(b2, b2p, NLAY, EMB, CP);
    pad_bias<<<(CP + 255) / 256, 256, 0, stream>>>(pb, pbp, 1, EMB, CP);
  }

  const int nodeThreads = M * CP;
  const int edgeThreads = E * 75;
  unsigned short* fbf[2] = { fbf0, fbf1 };

  for (int b = 0; b < 2; ++b) {
    embed_kernel<<<(nodeThreads + 255) / 256, 256, 0, stream>>>(xs[b], ae1, ae2, h, M);

    for (int l = 0; l < NLAY; ++l) {
      const float* ee1l = ee1 + (size_t)l * 6 * EMB;
      const float* ee2l = ee2 + (size_t)l * 3 * EMB;

      hipMemsetAsync(agg, 0, (size_t)MP * CP * 4, stream);
      edge_scatter<<<(edgeThreads + 255) / 256, 256, 0, stream>>>(
          h, eis[b], eas[b], ee1l, ee2l, agg, E);
      add_self_bf<<<(nodeThreads + 255) / 256, 256, 0, stream>>>(
          h, ee1l, ee2l, agg, aggbf, M);

      // MLP: mid = relu(agg @ W1 + b1) ; h = mid @ W2 + b2   (in place over h)
      launch_gemm(aggbf, w1t + (size_t)l * HP * CP, b1p + (size_t)l * HP,
                  nullptr, midbf, M, MP, HP, CP, 1.f, 1, stream);
      launch_gemm(midbf, w2t + (size_t)l * CP * HP, b2p + (size_t)l * CP,
                  h, nullptr, M, MP, CP, HP, 1.f, 0, stream);

      hipMemsetAsync(stats, 0, (size_t)2 * CP * 4, stream);
      bn_stats<<<256, CP, 0, stream>>>(h, sums, sqs, M);
      bn_apply<<<(nodeThreads + 255) / 256, 256, 0, stream>>>(
          h, hbf, sums, sqs, gam + (size_t)l * EMB, bet + (size_t)l * EMB,
          M, (l != NLAY - 1) ? 1 : 0);
    }

    // projector -> agg (f32, reuse), then graph mean-pool + L2 normalize
    launch_gemm(hbf, pjt, pbp, agg, nullptr, M, MP, CP, CP, 1.f, 0, stream);

    hipMemsetAsync(pooled, 0, ((size_t)NG * CP + NG) * 4, stream);
    pool_sum<<<(M * 75 + 255) / 256, 256, 0, stream>>>(agg, bts[b], pooled, M);
    pool_count<<<(M + 255) / 256, 256, 0, stream>>>(bts[b], cnt, M);
    pool_norm<<<NG, CP, 0, stream>>>(pooled, cnt, fbf[b]);
  }

  // logits = (f0 @ f1^T) / TEMP   -- Bt layout [N][K] == f1 itself
  launch_gemm(fbf0, fbf1, nullptr, (float*)d_out, nullptr,
              NG, NG, NG, CP, 25.0f /* 1/0.04 */, 0, stream);
}
